// SpikeBudgetSNN_4913442587075
// MI455X (gfx1250) — matrix operand order repaired and checked by reading, not compiled, not run on hardware
//
#include <hip/hip_runtime.h>

typedef __attribute__((ext_vector_type(16))) _Float16 v16h;
typedef __attribute__((ext_vector_type(8)))  _Float16 v8h;
typedef __attribute__((ext_vector_type(8)))  float    v8f;
typedef int v4i __attribute__((vector_size(16)));
typedef __attribute__((address_space(1))) v4i* as1_v4i;
typedef __attribute__((address_space(3))) v4i* as3_v4i;

#define HIDDEN 800
#define BATCH  512
#define TSTEPS 50
#define INDIM  784
#define KPAD   800   // all GEMM K padded to 800 (multiple of 32) at convert time
#define NPADB  896   // B (weight) rows padded to 7*128 so staging needs no guard
#define BETA   0.95f
#define LDSS   40    // LDS row stride in halves (80B, 16B-aligned, conflict-free)
#define NCH    (KPAD / 32)   // 25 K-chunks

#if defined(__HIP_DEVICE_COMPILE__) && __has_builtin(__builtin_amdgcn_global_load_async_to_lds_b128)
#define HAVE_ASYNC 1
#else
#define HAVE_ASYNC 0
#endif

// ---------------- f32 -> f16 conversion with row + K zero-padding -----------
// dst is [rowsDst][KPAD]; dst[r][c] = (r<rowsSrc && c<Ksrc) ? (f16)src[r][c] : 0
__global__ __launch_bounds__(256) void cvt_pad_f16(
    const float* __restrict__ src, _Float16* __restrict__ dst,
    int rowsSrc, int rowsDst, int Ksrc)
{
  int idx = blockIdx.x * 256 + threadIdx.x;
  if (idx < rowsDst * KPAD) {
    int r = idx / KPAD, c = idx - r * KPAD;
    dst[idx] = (r < rowsSrc && c < Ksrc) ? (_Float16)src[(size_t)r * Ksrc + c]
                                         : (_Float16)0.f;
  }
}

// ---------------- init: thresholds + zero states + zero s3sum ----------------
__global__ __launch_bounds__(256) void snn_init(
    const float* __restrict__ budget1, const float* __restrict__ budget2,
    float* __restrict__ thr1, float* __restrict__ thr2,
    float* __restrict__ m1, float* __restrict__ m2, float* __restrict__ m3,
    float* __restrict__ s3sum)
{
  int i = blockIdx.x * 256 + threadIdx.x;
  if (i < HIDDEN) {
    float c1 = fminf(fmaxf(budget1[i], 0.01f), 1.0f);
    float c2 = fminf(fmaxf(budget2[i], 0.01f), 1.0f);
    thr1[i] = 1.0f / c1;
    thr2[i] = 1.0f / c2;
  }
  if (i < BATCH * HIDDEN) { m1[i] = 0.0f; m2[i] = 0.0f; }
  if (i < BATCH * 16) m3[i] = 0.0f;
  if (i < BATCH * 10) s3sum[i] = 0.0f;
}

#if HAVE_ASYNC
// issue one 32-byte async copy (two b128) global -> LDS
__device__ __forceinline__ void stage_async32(const _Float16* gsrc, _Float16* ldst) {
  as1_v4i g = (as1_v4i)(uintptr_t)gsrc;
  as3_v4i l = (as3_v4i)(unsigned)(uintptr_t)ldst;
  __builtin_amdgcn_global_load_async_to_lds_b128(g, l, 0, 0);
  __builtin_amdgcn_global_load_async_to_lds_b128(g, l, 16, 0);
}
#endif

// ---------------- fused f16 GEMM + LIF epilogue ------------------------------
// A: [512, KPAD]  f16 row-major (zero-padded K)
// B: [NPADB, KPAD] f16 row-major (rows >= N are padding; columns >= N discarded)
// out[b,n] = spike( BETA*m + A@B^T + bias )
__global__ __launch_bounds__(256) void gemm_lif_spike(
    const _Float16* __restrict__ A16,
    const _Float16* __restrict__ B16,
    const float*    __restrict__ bias,
    const float*    __restrict__ thr,
    float*          __restrict__ mstate,   // [512, N] f32 (RMW)
    float*          __restrict__ s32,      // [512, N] f32 spike out (d_out)
    _Float16*       __restrict__ s16,      // [512, N] f16 spike out (next layer A)
    int N)
{
  __shared__ __align__(16) _Float16 As[2][128 * LDSS];   // double-buffered
  __shared__ __align__(16) _Float16 Bs[2][128 * LDSS];

  const int t     = threadIdx.x;
  const int lane  = t & 31;
  const int wave  = t >> 5;            // 0..7
  const int wm    = (wave >> 2) * 64;  // wave row base within tile
  const int wn    = (wave & 3) * 32;   // wave col base within tile
  const int lhalf = lane >> 4;         // 0/1 (lane group)
  const int l15   = lane & 15;
  const int blockN = blockIdx.x * 128;
  const int blockM = blockIdx.y * 128;

  const v8f vzero = {0.f, 0.f, 0.f, 0.f, 0.f, 0.f, 0.f, 0.f};
  v8f acc[4][2];
#pragma unroll
  for (int i = 0; i < 4; ++i)
#pragma unroll
    for (int j = 0; j < 2; ++j) acc[i][j] = vzero;

  // staging: 2 threads per row, 16 contiguous halves (2 x b128) each
  const int rs = t >> 1;          // 0..127
  const int ks = (t & 1) * 16;    // 0 or 16

  const _Float16* aS = A16 + (size_t)(blockM + rs) * KPAD + ks;  // rows < 512 always
  const _Float16* bS = B16 + (size_t)(blockN + rs) * KPAD + ks;  // rows < NPADB always
  _Float16* aD[2] = { &As[0][rs * LDSS + ks], &As[1][rs * LDSS + ks] };
  _Float16* bD[2] = { &Bs[0][rs * LDSS + ks], &Bs[1][rs * LDSS + ks] };

#if HAVE_ASYNC
  // prologue: chunk 0 -> buffer 0
  stage_async32(aS, aD[0]);
  stage_async32(bS, bD[0]);
#endif

  for (int i = 0; i < NCH; ++i) {
    const int cur = i & 1;

#if HAVE_ASYNC
    if (i + 1 < NCH) {
      // issue chunk i+1 into the other buffer, then wait only for chunk i
      stage_async32(aS + (i + 1) * 32, aD[cur ^ 1]);
      stage_async32(bS + (i + 1) * 32, bD[cur ^ 1]);
      __builtin_amdgcn_s_wait_asynccnt(4);   // in-order: chunk i's 4 loads done
    } else {
      __builtin_amdgcn_s_wait_asynccnt(0);
    }
#else
    {
      const v8h* sa = (const v8h*)(aS + i * 32);
      v8h a0 = sa[0], a1 = sa[1];
      *(v8h*)(aD[cur])     = a0;
      *(v8h*)(aD[cur] + 8) = a1;
      const v8h* sb = (const v8h*)(bS + i * 32);
      v8h b0 = sb[0], b1 = sb[1];
      *(v8h*)(bD[cur])     = b0;
      *(v8h*)(bD[cur] + 8) = b1;
    }
    if (i + 1 < NCH) {
      __builtin_prefetch(aS + (i + 1) * 32, 0, 1);
      __builtin_prefetch(bS + (i + 1) * 32, 0, 1);
    }
#endif

    __syncthreads();

    const _Float16* AsB = As[cur];
    const _Float16* BsB = Bs[cur];

    // ---- gather WMMA fragments per ISA wave32 layouts ----
    // A 16x32 f16: lanes 0-15 hold K=0..7 & 16..23, lanes 16-31 hold K=8..15 & 24..31
    v16h af[4];
#pragma unroll
    for (int mt = 0; mt < 4; ++mt) {
      const _Float16* row = AsB + (wm + mt * 16 + l15) * LDSS + lhalf * 8;
      v8h lo = *(const v8h*)row;          // K = lhalf*8 .. +7
      v8h hi = *(const v8h*)(row + 16);   // K = 16 + lhalf*8 .. +7
      af[mt] = __builtin_shufflevector(lo, hi, 0,1,2,3,4,5,6,7,8,9,10,11,12,13,14,15);
    }
    // B 32x16 f16: lane holds column n=l15; lanes 0-15 K=0..7 & 16..23, lanes 16-31 K=8..15 & 24..31
    v16h bf[2];
#pragma unroll
    for (int nt = 0; nt < 2; ++nt) {
      const _Float16* row = BsB + (wn + nt * 16 + l15) * LDSS + lhalf * 8;
      v8h lo = *(const v8h*)row;
      v8h hi = *(const v8h*)(row + 16);
      bf[nt] = __builtin_shufflevector(lo, hi, 0,1,2,3,4,5,6,7,8,9,10,11,12,13,14,15);
    }

#pragma unroll
    for (int mt = 0; mt < 4; ++mt)
#pragma unroll
      for (int nt = 0; nt < 2; ++nt)
        acc[mt][nt] = __builtin_amdgcn_wmma_f32_16x16x32_f16(
            false, af[mt], false, bf[nt], (short)0, acc[mt][nt], false, false);

    __syncthreads();   // all waves done reading buf[cur] before it is re-staged
  }

  // ---- fused LIF epilogue: m = BETA*m + y + b; s = m>thr; m *= (1-s) ----
  // D layout: VGPR v -> M = v + 8*lhalf ; N = l15
#pragma unroll
  for (int mt = 0; mt < 4; ++mt) {
#pragma unroll
    for (int nt = 0; nt < 2; ++nt) {
#pragma unroll
      for (int v = 0; v < 8; ++v) {
        int gm = blockM + wm + mt * 16 + v + lhalf * 8;
        int gn = blockN + wn + nt * 16 + l15;
        if (gn < N) {
          size_t off = (size_t)gm * N + gn;
          float y    = acc[mt][nt][v] + bias[gn];
          float mnew = BETA * mstate[off] + y;
          float s    = (mnew > thr[gn]) ? 1.0f : 0.0f;
          mstate[off] = mnew * (1.0f - s);
          s32[off] = s;
          s16[off] = (_Float16)s;
        }
      }
    }
  }
}

// ---------------- layer 3 (N=10): VALU dot + LIF + s3 accumulation ----------
__global__ __launch_bounds__(256) void layer3_spike(
    const _Float16* __restrict__ s2, const float* __restrict__ W3,
    const float* __restrict__ b3, float* __restrict__ m3,
    float* __restrict__ s3sum, float* __restrict__ s3r)
{
  int idx = blockIdx.x * 256 + threadIdx.x;
  if (idx >= BATCH * 10) return;
  int b = idx / 10, j = idx - b * 10;
  const _Float16* a = s2 + (size_t)b * HIDDEN;
  const float*    w = W3 + (size_t)j * HIDDEN;
  float acc = 0.f;
#pragma unroll 8
  for (int k = 0; k < HIDDEN; ++k) acc += (float)a[k] * w[k];
  int   mo   = b * 16 + j;
  float mnew = BETA * m3[mo] + acc + b3[j];
  float s    = (mnew > 1.0f) ? 1.0f : 0.0f;
  m3[mo]     = mnew * (1.0f - s);
  s3r[idx]   = s;
  s3sum[idx] += s;
}

// ---------------- host-side sequencing ----------------
extern "C" void kernel_launch(void* const* d_in, const int* in_sizes, int n_in,
                              void* d_out, int out_size, void* d_ws, size_t ws_size,
                              hipStream_t stream) {
  (void)in_sizes; (void)n_in; (void)out_size; (void)ws_size;
  const float* x       = (const float*)d_in[0];
  const float* W1      = (const float*)d_in[1];
  const float* b1      = (const float*)d_in[2];
  const float* W2      = (const float*)d_in[3];
  const float* b2      = (const float*)d_in[4];
  const float* W3      = (const float*)d_in[5];
  const float* b3      = (const float*)d_in[6];
  const float* budget1 = (const float*)d_in[7];
  const float* budget2 = (const float*)d_in[8];
  float* out = (float*)d_out;

  const size_t layerElems = (size_t)BATCH * HIDDEN;       // 409600 (HIDDEN==KPAD)
  const size_t xRows      = (size_t)TSTEPS * BATCH;       // 25600
  const size_t x16Elems   = xRows * KPAD;
  const size_t whElems    = (size_t)NPADB * KPAD;         // padded weight rows

  // workspace layout
  char* ws = (char*)d_ws;
  float* thr1 = (float*)ws;                 ws += HIDDEN * sizeof(float);
  float* thr2 = (float*)ws;                 ws += HIDDEN * sizeof(float);
  float* m1   = (float*)ws;                 ws += layerElems * sizeof(float);
  float* m2   = (float*)ws;                 ws += layerElems * sizeof(float);
  float* m3   = (float*)ws;                 ws += (size_t)BATCH * 16 * sizeof(float);
  _Float16* s1buf = (_Float16*)ws;          ws += layerElems * sizeof(_Float16);
  _Float16* s2buf = (_Float16*)ws;          ws += layerElems * sizeof(_Float16);
  _Float16* x16   = (_Float16*)ws;          ws += x16Elems * sizeof(_Float16);
  _Float16* W1h   = (_Float16*)ws;          ws += whElems * sizeof(_Float16);
  _Float16* W2h   = (_Float16*)ws;

  // one-time per launch: padded f16 copies of x, W1, W2 + state init
  cvt_pad_f16<<<(int)((x16Elems + 255) / 256), 256, 0, stream>>>(
      x, x16, (int)xRows, (int)xRows, INDIM);
  cvt_pad_f16<<<(int)((whElems + 255) / 256), 256, 0, stream>>>(
      W1, W1h, HIDDEN, NPADB, INDIM);
  cvt_pad_f16<<<(int)((whElems + 255) / 256), 256, 0, stream>>>(
      W2, W2h, HIDDEN, NPADB, HIDDEN);
  snn_init<<<(BATCH * HIDDEN + 255) / 256, 256, 0, stream>>>(
      budget1, budget2, thr1, thr2, m1, m2, m3, out);

  const size_t O1 = (size_t)BATCH * 10;                   // s1r block after s3sum
  const size_t O2 = O1 + (size_t)TSTEPS * layerElems;     // s2r block
  const size_t O3 = O2 + (size_t)TSTEPS * layerElems;     // s3r block

  dim3 gridG(NPADB / 128, BATCH / 128);  // 7 x 4
  for (int t = 0; t < TSTEPS; ++t) {
    gemm_lif_spike<<<gridG, 256, 0, stream>>>(
        x16 + (size_t)t * BATCH * KPAD, W1h, b1, thr1, m1,
        out + O1 + (size_t)t * layerElems, s1buf, HIDDEN);
    gemm_lif_spike<<<gridG, 256, 0, stream>>>(
        s1buf, W2h, b2, thr2, m2,
        out + O2 + (size_t)t * layerElems, s2buf, HIDDEN);
    layer3_spike<<<(BATCH * 10 + 255) / 256, 256, 0, stream>>>(
        s2buf, W3, b3, m3, out, out + O3 + (size_t)t * BATCH * 10);
  }
}
